// EdgeMLPDecoder_48086453846461
// MI455X (gfx1250) — matrix operand order repaired and checked by reading, not compiled, not run on hardware
//
#include <hip/hip_runtime.h>
#include <hip/hip_bf16.h>

typedef __attribute__((ext_vector_type(16))) _Float16 v16h;
typedef __attribute__((ext_vector_type(8)))  float    v8f;
typedef __attribute__((vector_size(16)))     int      vsi4;   // matches builtin param

#define DIM     128
#define HIDDEN  128
#define INDIM   512              // 4*DIM
#define KSTEPS  (INDIM / 32)     // 16 WMMA K-steps
#define NTILES  (HIDDEN / 16)    // 8 WMMA N-tiles

#if defined(__has_builtin)
#if __has_builtin(__builtin_amdgcn_global_load_async_to_lds_b128) && \
    __has_builtin(__builtin_amdgcn_s_wait_asynccnt)
#define HAVE_ASYNC_LDS 1
#endif
#endif

// ---------------------------------------------------------------------------
// Prep: W1 (512x128 f32, row-major [k][n]) -> W1T (128x512 f16, [n][k])
// so each B-fragment lane load is a contiguous 32B run of K.
// ---------------------------------------------------------------------------
__global__ void prep_w1_f16t(const float* __restrict__ W1,
                             _Float16* __restrict__ W1T) {
    int idx = blockIdx.x * blockDim.x + threadIdx.x;
    if (idx < INDIM * HIDDEN) {
        int k = idx / HIDDEN;
        int n = idx % HIDDEN;
        W1T[n * INDIM + k] = (_Float16)W1[idx];
    }
}

// ---------------------------------------------------------------------------
// Main: one wave computes 16 edges as a 16x512 @ 512x128 WMMA GEMM (f16 in,
// f32 accumulate), then bias+ReLU+128->1 dot in registers.
// Block = 256 threads = 8 waves. W1T staged in LDS (128 KB of 320 KB/WGP).
// ---------------------------------------------------------------------------
__global__ void __launch_bounds__(256)
edge_mlp_wmma(const float* __restrict__ z,
              const long long* __restrict__ ei,   // [2, E] int64
              const _Float16* __restrict__ W1T,   // [HIDDEN][INDIM] f16
              const float* __restrict__ b1,       // [HIDDEN]
              const float* __restrict__ W2,       // [HIDDEN]
              const float* __restrict__ b2,       // [1]
              float* __restrict__ out,            // [E]
              int E, int numTiles) {
    extern __shared__ _Float16 lds_w1[];          // HIDDEN*INDIM f16 = 128 KB

    // Stage W1T into LDS once per workgroup (async DMA path on gfx1250).
    {
        const uint4* src = (const uint4*)W1T;
        uint4*       dst = (uint4*)lds_w1;
        const int total = (HIDDEN * INDIM * (int)sizeof(_Float16)) / 16; // 8192
#ifdef HAVE_ASYNC_LDS
        for (int i = threadIdx.x; i < total; i += blockDim.x)
            __builtin_amdgcn_global_load_async_to_lds_b128(
                (vsi4*)(src + i), (vsi4*)(dst + i), /*offset=*/0, /*cpol=*/0);
        __builtin_amdgcn_s_wait_asynccnt(0);
#else
        for (int i = threadIdx.x; i < total; i += blockDim.x) dst[i] = src[i];
#endif
    }
    __syncthreads();

    const int wave = threadIdx.x >> 5;
    const int lane = threadIdx.x & 31;
    const int tile = blockIdx.x * (blockDim.x >> 5) + wave;
    if (tile >= numTiles) return;                 // wave-uniform: EXEC stays all-1s

    const int m  = lane & 15;                     // A-row / C-column index
    const int hf = lane >> 4;                     // lane half
    const int k0 = hf * 16;                       // per-lane A K-offset

    int e = tile * 16 + m;
    int ec = e < E ? e : (E - 1);                 // clamp tail, keep EXEC full
    const long long u = ei[ec];
    const long long v = ei[(long long)E + ec];
    const float* __restrict__ zu = z + u * (long long)DIM;
    const float* __restrict__ zv = z + v * (long long)DIM;

    // Load this lane's 64 zu + 64 zv values ONCE (the per-lane K pattern is
    // identical in all four feature segments), convert to f16 registers.
    // Index j = sp*16 + r*8 + i  <->  embedding element 32*sp + k0 + 8*r + i.
    _Float16 zuh[64], zvh[64];
#pragma unroll
    for (int sp = 0; sp < 4; ++sp) {
#pragma unroll
        for (int r = 0; r < 2; ++r) {
            const int base = 32 * sp + 8 * r + k0;
            const int j    = sp * 16 + r * 8;
            float4 q0 = *(const float4*)(zu + base);
            float4 q1 = *(const float4*)(zu + base + 4);
            zuh[j + 0] = (_Float16)q0.x; zuh[j + 1] = (_Float16)q0.y;
            zuh[j + 2] = (_Float16)q0.z; zuh[j + 3] = (_Float16)q0.w;
            zuh[j + 4] = (_Float16)q1.x; zuh[j + 5] = (_Float16)q1.y;
            zuh[j + 6] = (_Float16)q1.z; zuh[j + 7] = (_Float16)q1.w;
            float4 r0 = *(const float4*)(zv + base);
            float4 r1 = *(const float4*)(zv + base + 4);
            zvh[j + 0] = (_Float16)r0.x; zvh[j + 1] = (_Float16)r0.y;
            zvh[j + 2] = (_Float16)r0.z; zvh[j + 3] = (_Float16)r0.w;
            zvh[j + 4] = (_Float16)r1.x; zvh[j + 5] = (_Float16)r1.y;
            zvh[j + 6] = (_Float16)r1.z; zvh[j + 7] = (_Float16)r1.w;
        }
    }

    // Per-lane slices of bias / W2 at hidden column N = n*16 + m.
    float b1v[NTILES], w2v[NTILES];
#pragma unroll
    for (int n = 0; n < NTILES; ++n) {
        b1v[n] = b1[n * 16 + m];
        w2v[n] = W2[n * 16 + m];
    }
    const float bias2 = b2[0];

    v8f acc[NTILES] = {};

    // Fully unrolled: seg and register indices are compile-time constants,
    // so the hot loop is pure f16 ALU + ds_load_b128 + v_wmma.
#pragma unroll
    for (int s = 0; s < KSTEPS; ++s) {
        const int seg = s >> 2;                   // feature segment
        const int sp  = s & 3;                    // step within segment
        v16h a;
#pragma unroll
        for (int idx = 0; idx < 16; ++idx) {
            const int j = sp * 16 + idx;
            _Float16 x;
            if (seg == 0)      x = zuh[j];
            else if (seg == 1) x = zvh[j];
            else if (seg == 2) x = (_Float16)(zuh[j] * zvh[j]);
            else {
                _Float16 d = (_Float16)(zuh[j] - zvh[j]);
                x = d >= (_Float16)0 ? d : (_Float16)(-d);
            }
            a[idx] = x;
        }

        // B fragments from LDS: lane holds column N=n*16+m, K run
        // [32s + hf*16, +16) contiguous (32B, aligned) -> 8 WMMAs per K-step.
#pragma unroll
        for (int n = 0; n < NTILES; ++n) {
            const v16h bfrag =
                *(const v16h*)&lds_w1[(n * 16 + m) * INDIM + 32 * s + hf * 16];
            acc[n] = __builtin_amdgcn_wmma_f32_16x16x32_f16(
                false, a, false, bfrag, (short)0, acc[n], false, false);
        }
    }

    // Layer 2: per C-row r, lane holds h[N=n*16+m]; partial dot with W2,
    // then 16-lane butterfly reduce within each half-wave.
    float p[8];
#pragma unroll
    for (int r = 0; r < 8; ++r) {
        float sum = 0.0f;
#pragma unroll
        for (int n = 0; n < NTILES; ++n) {
            float h = acc[n][r] + b1v[n];
            h = h > 0.0f ? h : 0.0f;
            sum += h * w2v[n];
        }
        p[r] = sum;
    }
#pragma unroll
    for (int r = 0; r < 8; ++r) {
#pragma unroll
        for (int off = 1; off < 16; off <<= 1)
            p[r] += __shfl_xor(p[r], off, 32);
    }
    // C-row r of this lane-half is edge M = hf*8 + r; spread stores across lanes.
#pragma unroll
    for (int r = 0; r < 8; ++r) {
        const int eo = tile * 16 + hf * 8 + r;
        if (m == r && eo < E) out[eo] = p[r] + bias2;
    }
}

// ---------------------------------------------------------------------------
extern "C" void kernel_launch(void* const* d_in, const int* in_sizes, int n_in,
                              void* d_out, int out_size, void* d_ws, size_t ws_size,
                              hipStream_t stream) {
    const float*     z  = (const float*)d_in[0];
    const long long* ei = (const long long*)d_in[1];
    const float*     W1 = (const float*)d_in[2];
    const float*     b1 = (const float*)d_in[3];
    const float*     W2 = (const float*)d_in[4];
    const float*     b2 = (const float*)d_in[5];
    float*           out = (float*)d_out;

    const int E = in_sizes[1] / 2;

    _Float16* W1T = (_Float16*)d_ws;  // 512*128*2 = 128 KB scratch

    prep_w1_f16t<<<(INDIM * HIDDEN + 255) / 256, 256, 0, stream>>>(W1, W1T);

    const int numTiles = (E + 15) / 16;           // 31250 for E=500000
    const int wavesPerBlock = 8;                  // 256 threads, wave32
    const int blocks = (numTiles + wavesPerBlock - 1) / wavesPerBlock;
    const size_t shmem = (size_t)HIDDEN * INDIM * sizeof(_Float16); // 128 KB

    edge_mlp_wmma<<<blocks, 256, shmem, stream>>>(z, ei, W1T, b1, W2, b2, out,
                                                  E, numTiles);
}
